// CorrelationLayer_30623116820525
// MI455X (gfx1250) — compile-verified, matrix operand order checked by reading, not run on hardware
//
#include <hip/hip_runtime.h>
#include <hip/hip_bf16.h>

// CorrelationLayer for MI455X (gfx1250, wave32, WMMA).
//
// Roofline: 2.72 GFLOP vs ~155 MB min HBM traffic @ 23.3 TB/s => need ~400+
// TFLOP/s of dot-product throughput to stay memory bound -> v_wmma f16/f32.
//
//   K0: hipMemsetAsync zeroes the padded fB workspace (zero halo == the
//       reference's zero padding; no bounds checks in the hot loop).
//   K1: L2-normalize (fp32 math), convert to f16, transpose BCHW -> BHWC.
//       fA -> dense (B,64,64,256); fB -> padded (B,72,80,256), interior at
//       (y+4, x+4). Both resident in the 192 MB L2 for kernel 2's 9x reuse.
//   K2: one wave per 16-pixel A tile. A operand (8 K-steps) loaded once and
//       held in VGPRs. For each of 9 vertical displacements, two 16x16 Gram
//       tiles (B tiles at padded x0 and x0+16) accumulated over K=256 via
//       16x v_wmma_f32_16x16x32_f16; B operands software-pipelined one
//       K-step ahead so loads overlap the previous WMMA pair. The 9
//       horizontal displacements are the banded diagonals Gram[i][i+ox],
//       extracted via a 2 KB LDS bounce using all 32 lanes.

typedef _Float16 v16h __attribute__((ext_vector_type(16)));
typedef _Float16 h8   __attribute__((ext_vector_type(8)));
typedef _Float16 h2   __attribute__((ext_vector_type(2)));
typedef float    v8f  __attribute__((ext_vector_type(8)));

#define B_   16
#define C_   256
#define H_   64
#define W_   64
#define D_   4
#define KD_  81            // (2*4+1)^2 displacements
#define NK_  (C_/32)       // 8 K-steps per Gram tile
#define HW_  (H_*W_)
#define HP_  (H_ + 2*D_)   // 72  padded height (4 top, 4 bottom)
#define WP_  (W_ + 16)     // 80  padded width  (4 left, 12 right: tile1 reads to x0+31)

union V16 { v16h v; h8 h[2]; };

// ---------------------------------------------------------------------------
// Kernel 1: per-pixel L2 normalize + fp32->f16 + BCHW -> pixel-major.
// One block = one image row (b,y); 64 threads = 64 x-pixels. Channel-plane
// reads are fully coalesced; f16 stores are per-thread sequential within a
// private 512B region -> fully-written L2 lines, no HBM write amplification.
// ---------------------------------------------------------------------------
__global__ void __launch_bounds__(64)
corr_normalize_f16(const float* __restrict__ fA,
                   const float* __restrict__ fB,
                   _Float16* __restrict__ wsA,
                   _Float16* __restrict__ wsB) {
    const int x = threadIdx.x;   // 0..63
    const int y = blockIdx.x;    // 0..63
    const int b = blockIdx.y;    // 0..15
    const bool isB = (blockIdx.z != 0);
    const float* __restrict__ src = isB ? fB : fA;

    const size_t base = (size_t)b * (C_*HW_) + (size_t)y * W_ + x;

    float s = 0.0f;
    #pragma unroll 8
    for (int c = 0; c < C_; ++c) {
        float v = src[base + (size_t)c * HW_];
        s = fmaf(v, v, s);
    }
    // matches torch F.normalize: x / max(||x||, 1e-12)
    const float inv = 1.0f / fmaxf(sqrtf(s), 1e-12f);

    _Float16* __restrict__ dst;
    size_t obase;
    if (isB) {   // padded layout, interior at (y+4, x+4)
        dst = wsB;
        obase = (size_t)((b*HP_ + (y + D_))*WP_ + (x + D_)) * C_;
    } else {     // dense layout
        dst = wsA;
        obase = (size_t)(b*HW_ + y*W_ + x) * C_;
    }

    #pragma unroll 4
    for (int c = 0; c < C_; c += 2) {
        float v0 = src[base + (size_t)(c    ) * HW_] * inv;
        float v1 = src[base + (size_t)(c + 1) * HW_] * inv;
        h2 p;
        p.x = (_Float16)v0;
        p.y = (_Float16)v1;
        *(h2*)(dst + obase + c) = p;
    }
}

// ---------------------------------------------------------------------------
// Kernel 2: WMMA correlation. One wave per block (wave32).
// ---------------------------------------------------------------------------
__device__ inline V16 ld16(const _Float16* p, int koff0, int koff1) {
    V16 r;
    r.h[0] = *(const h8*)(p + koff0);
    r.h[1] = *(const h8*)(p + koff1);
    return r;
}

__global__ void __launch_bounds__(32)
corr_wmma(const _Float16* __restrict__ A16,
          const _Float16* __restrict__ B16,
          float* __restrict__ out) {
    const int lane = threadIdx.x;
    const int g = lane >> 4;     // K-half selector for A/B operand layout
    const int n = lane & 15;     // M (A pixel) / N (B pixel) index
    const int x0 = blockIdx.x * 16;
    const int y  = blockIdx.y;
    const int b  = blockIdx.z;

    __shared__ float gram[2][16][16];

    // v16h per-lane layout: lane g=lane>>4 holds K = kk+8g.. and kk+16+8g..
    const int koff0 = g * 8;
    const int koff1 = 16 + g * 8;

    // A tile: pixels (y, x0..x0+15); load all 8 K-steps once, hold in VGPRs.
    const _Float16* aptr = A16 + (size_t)((b*H_ + y)*W_ + (x0 + n)) * C_;
    V16 a[NK_];
    #pragma unroll
    for (int s = 0; s < NK_; ++s)
        a[s] = ld16(aptr + s*32, koff0, koff1);

    for (int oy = 0; oy < 2*D_ + 1; ++oy) {
        // Padded B row y+oy; tile0 at padded x0+n, tile1 at +16 pixels
        // (= +8192 B -> immediate offsets off one base pointer).
        const _Float16* bptr =
            B16 + (size_t)((b*HP_ + (y + oy))*WP_ + (x0 + n)) * C_;

        v8f acc0 = {};
        v8f acc1 = {};

        // Software-pipelined B stream: loads for step s+1 issue before the
        // WMMAs of step s consume their data.
        V16 b0 = ld16(bptr,          koff0, koff1);
        V16 b1 = ld16(bptr + 16*C_,  koff0, koff1);
        #pragma unroll
        for (int s = 0; s < NK_; ++s) {
            V16 nb0, nb1;
            if (s + 1 < NK_) {
                nb0 = ld16(bptr + (s+1)*32,         koff0, koff1);
                nb1 = ld16(bptr + 16*C_ + (s+1)*32, koff0, koff1);
            }
            acc0 = __builtin_amdgcn_wmma_f32_16x16x32_f16(
                false, a[s].v, false, b0.v, (short)0, acc0, false, false);
            acc1 = __builtin_amdgcn_wmma_f32_16x16x32_f16(
                false, a[s].v, false, b1.v, (short)0, acc1, false, false);
            if (s + 1 < NK_) { b0 = nb0; b1 = nb1; }
        }

        // C/D layout: VGPR r holds M = r + 8*(lane>>4), N = lane&15.
        #pragma unroll
        for (int r = 0; r < 8; ++r) {
            gram[0][8*g + r][n] = acc0[r];
            gram[1][8*g + r][n] = acc1[r];
        }
        __syncthreads();

        // Banded-diagonal extraction: out(i, ox) = Gram[i][i+ox], ox = 0..8.
        // Lanes 0-15 handle even ox, lanes 16-31 odd ox.
        {
            const int i = n;
            #pragma unroll
            for (int t = 0; t < 5; ++t) {
                const int ox = 2*t + g;
                if (ox < 2*D_ + 1) {
                    const int jj = i + ox;                    // 0..23
                    const float val = (jj < 16) ? gram[0][i][jj]
                                                : gram[1][i][jj - 16];
                    const int k = oy * (2*D_ + 1) + ox;
                    out[(((size_t)b*KD_ + k)*H_ + y)*W_ + (x0 + i)] = val;
                }
            }
        }
        __syncthreads();   // protect gram before next oy overwrites it
    }
}

// ---------------------------------------------------------------------------
extern "C" void kernel_launch(void* const* d_in, const int* in_sizes, int n_in,
                              void* d_out, int out_size, void* d_ws, size_t ws_size,
                              hipStream_t stream) {
    (void)in_sizes; (void)n_in; (void)out_size; (void)ws_size;
    const float* fA = (const float*)d_in[0];
    const float* fB = (const float*)d_in[1];
    float* out = (float*)d_out;

    // workspace: dense f16 fA (33.6 MB) + padded f16 fB (47.2 MB).
    _Float16* wsA = (_Float16*)d_ws;
    _Float16* wsB = wsA + (size_t)B_ * HW_ * C_;
    const size_t wsB_bytes = (size_t)B_ * HP_ * WP_ * C_ * sizeof(_Float16);

    hipMemsetAsync(wsB, 0, wsB_bytes, stream);   // zero halo == zero padding
    corr_normalize_f16<<<dim3(H_, B_, 2), 64, 0, stream>>>(fA, fB, wsA, wsB);
    corr_wmma<<<dim3(W_/16, H_, B_), 32, 0, stream>>>(wsA, wsB, out);
}
